// TLayer_76252849373352
// MI455X (gfx1250) — matrix operand order repaired and checked by reading, not compiled, not run on hardware
//
#include <hip/hip_runtime.h>
#include <hip/hip_bf16.h>

// ---------------------------------------------------------------------------
// Transformer layer for MI455X (gfx1250), bf16 WMMA path + TDM A-tile staging.
//   D=1024, NH=16, DH=64, DM=4096, B=2, S=2048  -> rows = B*S = 4096
// ---------------------------------------------------------------------------

typedef __bf16 bf16;
typedef __attribute__((ext_vector_type(16))) __bf16    v16bf;
typedef __attribute__((ext_vector_type(8)))  float     v8f;
typedef __attribute__((ext_vector_type(4)))  unsigned  v4u;
typedef __attribute__((ext_vector_type(8)))  unsigned  v8u;

#define D_MODEL 1024
#define N_HEADS 16
#define D_HEAD  64
#define D_MLP   4096
#define BATCH   2
#define SEQ     2048
#define ROWS    (BATCH * SEQ)   // 4096

// ----------------------------- WMMA helpers --------------------------------

__device__ __forceinline__ v8f wmma_bf16(v16bf a, v16bf b, v8f c) {
  return __builtin_amdgcn_wmma_f32_16x16x32_bf16(
      /*neg_a=*/false, a, /*neg_b=*/false, b,
      /*c_mod=*/(short)0, c, /*reuse_a=*/false, /*reuse_b=*/false);
}

// A fragment (16x32 bf16) from row-major storage, ld in elements.
// row = lane&15; per lane K = koff..koff+15 contiguous (koff = 0 or 16).
__device__ __forceinline__ v16bf load_a_frag(const bf16* __restrict__ A, int lda, int lane) {
  int row  = lane & 15;
  int koff = (lane & 16) ? 16 : 0;
  const bf16* p = A + (size_t)row * lda + koff;
  union { v16bf v; uint4 q[2]; } r;
  r.q[0] = *(const uint4*)p;
  r.q[1] = *(const uint4*)(p + 8);
  return r.v;
}

// B fragment (32x16) from memory holding B^T rows contiguously:
// element (k,n) at p[n*ld + k]; per lane K = koff..koff+15 contiguous.
__device__ __forceinline__ v16bf load_bt_frag(const bf16* __restrict__ Bt, int ld, int lane) {
  int n    = lane & 15;
  int koff = (lane & 16) ? 16 : 0;
  const bf16* p = Bt + (size_t)n * ld + koff;
  union { v16bf v; uint4 q[2]; } r;
  r.q[0] = *(const uint4*)p;
  r.q[1] = *(const uint4*)(p + 8);
  return r.v;
}

// ----------------------------- TDM (tensor DMA) -----------------------------
// Issue TENSOR_LOAD_TO_LDS for a 2D bf16 tile: tile_dim0=32 elems (K),
// tile_dim1=128 rows (M), tensor row stride = lda elems. 2-group descriptor
// (groups 2/3 NULL -> up-to-2D tensor). Fields per CDNA5 ISA.
__device__ __forceinline__ void tdm_load_a_tile(const bf16* gaddr, unsigned lds_off,
                                                unsigned dim0, unsigned dim1,
                                                unsigned stride0) {
  unsigned long long ga = (unsigned long long)(size_t)gaddr;
  v4u g0;
  g0[0] = 1u;                                              // count=1, user desc
  g0[1] = lds_off;                                         // lds_addr (bytes)
  g0[2] = (unsigned)ga;                                    // global_addr[31:0]
  g0[3] = (unsigned)((ga >> 32) & 0x01FFFFFFu) | (2u << 30); // addr[56:32] | type=2
  v8u g1;
  g1[0] = 1u << 16;                                        // data_size=1 (2 bytes)
  g1[1] = (dim0 & 0xFFFFu) << 16;                          // tensor_dim0[15:0]
  g1[2] = ((dim0 >> 16) & 0xFFFFu) | ((dim1 & 0xFFFFu) << 16); // dim0 hi | dim1 lo
  g1[3] = ((dim1 >> 16) & 0xFFFFu) | (32u << 16);          // dim1 hi | tile_dim0=32
  g1[4] = 128u;                                            // tile_dim1=128, tile_dim2=0
  g1[5] = stride0;                                         // tensor_dim0_stride[31:0]
  g1[6] = 0u;                                              // stride hi | dim1_stride lo
  g1[7] = 0u;
  asm volatile("tensor_load_to_lds %0, %1" :: "s"(g0), "s"(g1) : "memory");
}

// ----------------------------- weight prep ---------------------------------

__global__ __launch_bounds__(256) void cvt_bf16_kernel(const float* __restrict__ in,
                                                       bf16* __restrict__ out, int n) {
  int i = blockIdx.x * 256 + threadIdx.x;
  if (i < n) out[i] = (bf16)in[i];
}

// W[NH, D, DH] -> out[D, NH*DH] row-major bf16
__global__ __launch_bounds__(256) void repack_qkv_kernel(const float* __restrict__ W,
                                                         bf16* __restrict__ out) {
  int i = blockIdx.x * 256 + threadIdx.x;
  int n = i & (D_MODEL - 1);
  int d = i >> 10;
  int h = n >> 6;
  int e = n & 63;
  out[i] = (bf16)W[((size_t)(h * D_MODEL + d) << 6) + e];
}

// ----------------------------- layernorm -----------------------------------

__global__ __launch_bounds__(256) void ln_bf16_kernel(const float* __restrict__ x,
                                                      const float* __restrict__ w,
                                                      const float* __restrict__ b,
                                                      bf16* __restrict__ out) {
  __shared__ float red[256];
  int row = blockIdx.x;
  const float* xr = x + (size_t)row * D_MODEL;
  float lx[4];
  float s = 0.f;
#pragma unroll
  for (int i = 0; i < 4; ++i) { lx[i] = xr[threadIdx.x + 256 * i]; s += lx[i]; }
  red[threadIdx.x] = s;
  __syncthreads();
  for (int off = 128; off > 0; off >>= 1) {
    if (threadIdx.x < off) red[threadIdx.x] += red[threadIdx.x + off];
    __syncthreads();
  }
  float mu = red[0] * (1.f / D_MODEL);
  __syncthreads();
  float vs = 0.f;
#pragma unroll
  for (int i = 0; i < 4; ++i) { float d = lx[i] - mu; vs += d * d; }
  red[threadIdx.x] = vs;
  __syncthreads();
  for (int off = 128; off > 0; off >>= 1) {
    if (threadIdx.x < off) red[threadIdx.x] += red[threadIdx.x + off];
    __syncthreads();
  }
  float rstd = rsqrtf(red[0] * (1.f / D_MODEL) + 1e-5f);
#pragma unroll
  for (int i = 0; i < 4; ++i) {
    int c = threadIdx.x + 256 * i;
    out[(size_t)row * D_MODEL + c] = (bf16)((lx[i] - mu) * rstd * w[c] + b[c]);
  }
}

// ----------------------------- generic WMMA GEMM ---------------------------
// C[M,N] = A[M,K] @ B[K,N], both bf16 row-major. Block = 8 waves, tile
// 128(M) x 64(N). Per K-step of 32:
//   - wave 0 issues TDM copy of the 128x32 A tile into LDS (tensor_load_to_lds)
//   - all threads cooperatively load B tile (32x64) and transpose into LDS
//   - fragments read back as 2x ds_load_b128 each
// Dynamic LDS: [0,8192) A tile row-major [128][32]; [8192,..) Bt [64][48].

#define BT_LD 48

__global__ __launch_bounds__(256)
void gemm_bf16_kernel(const bf16* __restrict__ A, int lda,
                      const bf16* __restrict__ B, int ldb,
                      int M, int N, int K,
                      const float* __restrict__ bias,
                      const float* __restrict__ resid,
                      int act,
                      float* __restrict__ outF,
                      bf16* __restrict__ outH,
                      bf16* __restrict__ outHT) {   // outHT: [N][M]
  extern __shared__ bf16 smem[];
  bf16* At = smem;                 // 128*32 bf16 = 8192 B, LDS offset 0
  bf16* Bt = smem + 128 * 32;      // 64*BT_LD bf16

  int lane = threadIdx.x & 31;
  int wid  = threadIdx.x >> 5;
  int tiles_n = N >> 6;
  int bm = (blockIdx.x / tiles_n) * 128;
  int tn = (blockIdx.x % tiles_n) * 64;
  int tm = bm + wid * 16;

  v8f zero = {0.f, 0.f, 0.f, 0.f, 0.f, 0.f, 0.f, 0.f};
  v8f acc[4] = {zero, zero, zero, zero};

  const bf16* Ablk = A + (size_t)bm * lda;

  // cooperative B-tile coords: 256 threads cover 32 rows x 64 cols (8/thread)
  int kr = threadIdx.x >> 3;
  int c8 = (threadIdx.x & 7) << 3;

  for (int k0 = 0; k0 < K; k0 += 32) {
    __syncthreads();                     // previous tile fully consumed
    if (wid == 0)
      tdm_load_a_tile(Ablk + k0, 0u, (unsigned)lda, (unsigned)M, (unsigned)lda);
    {
      const bf16* src = B + (size_t)(k0 + kr) * ldb + tn + c8;
      union { uint4 q; bf16 h[8]; } u;
      u.q = *(const uint4*)src;                           // global_load_b128
#pragma unroll
      for (int j = 0; j < 8; ++j) Bt[(c8 + j) * BT_LD + kr] = u.h[j];
      if (k0 + 32 < K)
        __builtin_prefetch(B + (size_t)(k0 + 32 + kr) * ldb + tn + c8, 0, 0);
    }
    if (wid == 0) __builtin_amdgcn_s_wait_tensorcnt(0);   // TDM done before barrier
    __syncthreads();

    // A fragment from LDS A tile (row-major [128][32])
    v16bf a;
    {
      int arow  = wid * 16 + (lane & 15);
      int akoff = (lane & 16) ? 16 : 0;
      const bf16* pa = At + arow * 32 + akoff;
      union { v16bf v; uint4 q[2]; } r;
      r.q[0] = *(const uint4*)pa;                         // ds_load_b128
      r.q[1] = *(const uint4*)(pa + 8);                   // ds_load_b128
      a = r.v;
    }

    int n    = lane & 15;
    int koff = (lane & 16) ? 16 : 0;
#pragma unroll
    for (int c = 0; c < 4; ++c) {
      const bf16* p = &Bt[(c * 16 + n) * BT_LD + koff];
      union { v16bf v; uint4 q[2]; } r;
      r.q[0] = *(const uint4*)p;                          // ds_load_b128
      r.q[1] = *(const uint4*)(p + 8);                    // ds_load_b128
      acc[c] = wmma_bf16(a, r.v, acc[c]);
    }
  }

  int n_lo = lane & 15;
  int mo   = (lane & 16) ? 8 : 0;
#pragma unroll
  for (int c = 0; c < 4; ++c) {
#pragma unroll
    for (int r = 0; r < 8; ++r) {
      int row = tm + mo + r;
      int col = tn + c * 16 + n_lo;
      float v = acc[c][r];
      if (bias)  v += bias[col];
      if (act)   v = 0.5f * v * (1.0f + erff(v * 0.70710678118654752f)); // exact GELU
      size_t oi = (size_t)row * N + col;
      if (resid) v += resid[oi];
      if (outF)  outF[oi] = v;
      if (outH)  outH[oi] = (bf16)v;
      if (outHT) outHT[(size_t)col * M + row] = (bf16)v;
    }
  }
}

// ----------------------------- flash attention -----------------------------
// One wave = 16 query rows of one (b, h). Streaming softmax over 32-key chunks.
// K is [token][e] (contiguous along e); V pre-transposed Vt[e][token].

__global__ __launch_bounds__(128)
void attn_kernel(const bf16* __restrict__ Q, const bf16* __restrict__ Km,
                 const bf16* __restrict__ Vt, bf16* __restrict__ ctx) {
  __shared__ float s_sc[4][16][32];
  __shared__ bf16  s_p[4][16][32];
  __shared__ float s_stat[4][2][16];   // [0]=corr, [1]=l

  int lane = threadIdx.x & 31;
  int wid  = threadIdx.x >> 5;
  int tile = blockIdx.x * 4 + wid;
  const int QT = SEQ / 16;
  int b   = tile / (N_HEADS * QT);
  int rem = tile % (N_HEADS * QT);
  int h   = rem / QT;
  int qt  = rem % QT;
  int q0  = qt * 16;
  int row_base = b * SEQ + q0;
  const int ld = N_HEADS * D_HEAD;

  v16bf qf0 = load_a_frag(Q + (size_t)row_base * ld + h * D_HEAD,      ld, lane);
  v16bf qf1 = load_a_frag(Q + (size_t)row_base * ld + h * D_HEAD + 32, ld, lane);

  v8f zero = {0.f, 0.f, 0.f, 0.f, 0.f, 0.f, 0.f, 0.f};
  v8f o[4] = {zero, zero, zero, zero};
  float m_reg = -1e30f, l_reg = 0.f;

  int n_lo = lane & 15;
  int mo   = (lane & 16) ? 8 : 0;
  int koff16 = (lane & 16) ? 16 : 0;
  int nchunks = (q0 + 15) / 32 + 1;

  for (int c = 0; c < nchunks; ++c) {
    int k0 = c * 32;
    v8f s0 = zero, s1 = zero;
    {
      const bf16* kbase = Km + (size_t)(b * SEQ + k0) * ld + h * D_HEAD;
      v16bf kb00 = load_bt_frag(kbase,                ld, lane);
      v16bf kb10 = load_bt_frag(kbase + 16 * ld,      ld, lane);
      v16bf kb01 = load_bt_frag(kbase + 32,           ld, lane);
      v16bf kb11 = load_bt_frag(kbase + 16 * ld + 32, ld, lane);
      s0 = wmma_bf16(qf0, kb00, s0);
      s1 = wmma_bf16(qf0, kb10, s1);
      s0 = wmma_bf16(qf1, kb01, s0);
      s1 = wmma_bf16(qf1, kb11, s1);
    }
#pragma unroll
    for (int r = 0; r < 8; ++r) {
      s_sc[wid][mo + r][n_lo]      = s0[r];
      s_sc[wid][mo + r][16 + n_lo] = s1[r];
    }
    if (lane < 16) {
      int qg = q0 + lane;
      float mx = m_reg;
      float vals[32];
#pragma unroll
      for (int j = 0; j < 32; ++j) {
        float sv = s_sc[wid][lane][j] * 0.125f;     // 1/sqrt(64)
        sv = (k0 + j <= qg) ? sv : -1e30f;          // causal mask
        vals[j] = sv;
        mx = fmaxf(mx, sv);
      }
      float corr = __expf(m_reg - mx);
      float l = l_reg * corr;
#pragma unroll
      for (int j = 0; j < 32; ++j) {
        float p = __expf(vals[j] - mx);
        l += p;
        s_p[wid][lane][j] = (bf16)p;
      }
      m_reg = mx; l_reg = l;
      s_stat[wid][0][lane] = corr;
      s_stat[wid][1][lane] = l;
    }
#pragma unroll
    for (int r = 0; r < 8; ++r) {
      float f = s_stat[wid][0][mo + r];
      o[0][r] *= f; o[1][r] *= f; o[2][r] *= f; o[3][r] *= f;
    }
    v16bf pf;
    {
      int koff = (lane & 16) ? 16 : 0;
      const bf16* pp = &s_p[wid][lane & 15][koff];
      union { v16bf v; uint4 q[2]; } pr;
      pr.q[0] = *(const uint4*)pp;
      pr.q[1] = *(const uint4*)(pp + 8);
      pf = pr.v;
    }
#pragma unroll
    for (int ec = 0; ec < 4; ++ec) {
      const bf16* p = Vt + (size_t)(h * D_HEAD + ec * 16 + n_lo) * ROWS
                         + b * SEQ + k0 + koff16;
      union { v16bf v; uint4 q[2]; } r;
      r.q[0] = *(const uint4*)p;
      r.q[1] = *(const uint4*)(p + 8);
      o[ec] = wmma_bf16(pf, r.v, o[ec]);
    }
  }

#pragma unroll
  for (int r = 0; r < 8; ++r) {
    float linv = 1.f / s_stat[wid][1][mo + r];
    int rg = row_base + mo + r;
#pragma unroll
    for (int ec = 0; ec < 4; ++ec)
      ctx[(size_t)rg * ld + h * D_HEAD + ec * 16 + n_lo] = (bf16)(o[ec][r] * linv);
  }
}

// ----------------------------- host launcher -------------------------------

extern "C" void kernel_launch(void* const* d_in, const int* in_sizes, int n_in,
                              void* d_out, int out_size, void* d_ws, size_t ws_size,
                              hipStream_t stream) {
  (void)in_sizes; (void)n_in; (void)out_size; (void)ws_size;
  const float* x    = (const float*)d_in[0];
  const float* WQ   = (const float*)d_in[1];
  const float* WK   = (const float*)d_in[2];
  const float* WV   = (const float*)d_in[3];
  const float* WO   = (const float*)d_in[4];
  const float* W1   = (const float*)d_in[5];
  const float* b1   = (const float*)d_in[6];
  const float* W2   = (const float*)d_in[7];
  const float* b2   = (const float*)d_in[8];
  const float* ln1w = (const float*)d_in[9];
  const float* ln1b = (const float*)d_in[10];
  const float* ln2w = (const float*)d_in[11];
  const float* ln2b = (const float*)d_in[12];
  float* out = (float*)d_out;

  char* ws = (char*)d_ws;
  const size_t MB = 1024ull * 1024ull;
  bf16*  qb  = (bf16*)(ws);              //  8 MB   (overlaid by ub later)
  bf16*  kb  = (bf16*)(ws + 8 * MB);     //  8 MB
  bf16*  vt  = (bf16*)(ws + 16 * MB);    //  8 MB   V transposed: [NH*DH][ROWS]
  bf16*  ctx = (bf16*)(ws + 24 * MB);    //  8 MB
  bf16*  ub  = (bf16*)(ws);              // 32 MB MLP hidden, reuses q/k/v/ctx
  bf16*  hb  = (bf16*)(ws + 32 * MB);    //  8 MB   (LN1 out, then LN2 out)
  float* x1  = (float*)(ws + 40 * MB);   // 16 MB   residual stream after attn
  bf16*  wqr = (bf16*)(ws + 56 * MB);    //  2 MB
  bf16*  wkr = (bf16*)(ws + 58 * MB);    //  2 MB
  bf16*  wvr = (bf16*)(ws + 60 * MB);    //  2 MB
  bf16*  woc = (bf16*)(ws + 62 * MB);    //  2 MB
  bf16*  w1c = (bf16*)(ws + 64 * MB);    //  8 MB
  bf16*  w2c = (bf16*)(ws + 72 * MB);    //  8 MB   -> total 80 MB

  const size_t GEMM_LDS = (128 * 32 + 64 * BT_LD) * sizeof(bf16);  // 14336 B

  // --- weight prep (bf16) ---
  repack_qkv_kernel<<<4096, 256, 0, stream>>>(WQ, wqr);
  repack_qkv_kernel<<<4096, 256, 0, stream>>>(WK, wkr);
  repack_qkv_kernel<<<4096, 256, 0, stream>>>(WV, wvr);
  cvt_bf16_kernel<<<4096, 256, 0, stream>>>(WO, woc, D_MODEL * D_MODEL);
  cvt_bf16_kernel<<<16384, 256, 0, stream>>>(W1, w1c, D_MODEL * D_MLP);
  cvt_bf16_kernel<<<16384, 256, 0, stream>>>(W2, w2c, D_MLP * D_MODEL);

  // --- LN1 ---
  ln_bf16_kernel<<<ROWS, 256, 0, stream>>>(x, ln1w, ln1b, hb);

  // --- QKV projections: [4096,1024] @ [1024,1024] ---
  gemm_bf16_kernel<<<512, 256, GEMM_LDS, stream>>>(hb, D_MODEL, wqr, D_MODEL,
      ROWS, D_MODEL, D_MODEL, nullptr, nullptr, 0, nullptr, qb, nullptr);
  gemm_bf16_kernel<<<512, 256, GEMM_LDS, stream>>>(hb, D_MODEL, wkr, D_MODEL,
      ROWS, D_MODEL, D_MODEL, nullptr, nullptr, 0, nullptr, kb, nullptr);
  gemm_bf16_kernel<<<512, 256, GEMM_LDS, stream>>>(hb, D_MODEL, wvr, D_MODEL,
      ROWS, D_MODEL, D_MODEL, nullptr, nullptr, 0, nullptr, nullptr, vt); // V^T

  // --- causal flash attention ---
  attn_kernel<<<1024, 128, 0, stream>>>(qb, kb, vt, ctx);

  // --- output projection + residual: x1 = x + ctx @ WO ---
  gemm_bf16_kernel<<<512, 256, GEMM_LDS, stream>>>(ctx, D_MODEL, woc, D_MODEL,
      ROWS, D_MODEL, D_MODEL, nullptr, x, 0, x1, nullptr, nullptr);

  // --- LN2 ---
  ln_bf16_kernel<<<ROWS, 256, 0, stream>>>(x1, ln2w, ln2b, hb);

  // --- MLP1: GELU(h2 @ W1 + b1) -> bf16 [4096,4096] ---
  gemm_bf16_kernel<<<2048, 256, GEMM_LDS, stream>>>(hb, D_MODEL, w1c, D_MLP,
      ROWS, D_MLP, D_MODEL, b1, nullptr, 1, nullptr, ub, nullptr);

  // --- MLP2: out = x1 + u @ W2 + b2 ---
  gemm_bf16_kernel<<<512, 256, GEMM_LDS, stream>>>(ub, D_MLP, w2c, D_MODEL,
      ROWS, D_MODEL, D_MLP, b2, x1, 0, out, nullptr, nullptr);
}
